// GraphEncoder_36309653520480
// MI455X (gfx1250) — matrix operand order repaired and checked by reading, not compiled, not run on hardware
//
#include <hip/hip_runtime.h>
#include <hip/hip_bf16.h>

// ---------------------------------------------------------------------------
// GCN encoder for MI455X (gfx1250): bf16 WMMA GEMMs + L2 f32-atomic SpMM.
// ---------------------------------------------------------------------------

typedef __attribute__((ext_vector_type(16))) __bf16 v16bf;
typedef __attribute__((ext_vector_type(4)))  __bf16 v4bf;
typedef __attribute__((ext_vector_type(8)))  float  v8f;

// gfx1250 has native bf16: let the backend emit hardware f32->bf16 converts
// instead of a software round-to-nearest-even bit sequence.
static __device__ __forceinline__ __bf16 f2bf(float f) {
    return static_cast<__bf16>(f);
}

// ---------------------------------------------------------------------------
// Tiled GEMM:  C[M,NC] = op(A)[M,K] @ B[K,NC]  (+bias) (+relu)
//   op(A) = relu(A) if RELU_A (fuses previous layer's activation).
//   Block = 256 threads (8 waves). Block tile = 256 x NC.
//   Wave tile = 32 x NC (2 row-tiles -> each B fragment feeds 2 WMMAs).
//   K stepped by 32 using v_wmma_f32_16x16x32_bf16, f32 accumulation.
// ---------------------------------------------------------------------------
template <int NC, bool RELU_A, bool BIAS, bool RELU_OUT>
__global__ __launch_bounds__(256) void gemm_wmma_bf16(
    const float* __restrict__ A, const float* __restrict__ B,
    const float* __restrict__ bias, float* __restrict__ C,
    int M, int K)
{
    constexpr int BM = 256;
    constexpr int KB = 32;
    constexpr int NT = NC / 16;
    constexpr int MT = 2;

    // A tile stored per row in natural K order: [K0..7 | K8..15 | K16..23 | K24..31]
    // so each lane's half of the 16x32 bf16 A fragment is one contiguous 32B.
    __shared__ alignas(32) __bf16 sA[BM * KB];    // 16 KB
    // B tile stored N-major: sB[n*KB + k]; lane fragment = contiguous 16 K's.
    __shared__ alignas(32) __bf16 sB[NC * KB];    // <= 8 KB

    const int tid     = threadIdx.x;
    const int lane    = tid & 31;
    const int wave    = tid >> 5;
    const int rowBase = blockIdx.x * BM;

    const int m16   = lane & 15;   // row (A) / col (B,C) inside 16x16 tile
    const int khalf = lane >> 4;   // lane half selects K sub-range

    v8f acc[MT][NT];
#pragma unroll
    for (int mt = 0; mt < MT; ++mt)
#pragma unroll
        for (int t = 0; t < NT; ++t)
            acc[mt][t] = (v8f){0.f,0.f,0.f,0.f,0.f,0.f,0.f,0.f};

    for (int k0 = 0; k0 < K; k0 += KB) {
        // ---- stage A tile (BM x KB), f32 -> bf16, natural K order ----
#pragma unroll
        for (int i = 0; i < (BM * KB / 4) / 256; ++i) {
            int idx = tid + i * 256;          // float4 slot
            int r   = idx >> 3;               // KB/4 = 8 float4 per row
            int c4  = idx & 7;
            int gr  = rowBase + r;
            float4 v = make_float4(0.f, 0.f, 0.f, 0.f);
            if (gr < M) v = *(const float4*)(A + (size_t)gr * K + k0 + c4 * 4);
            if (RELU_A) {
                v.x = fmaxf(v.x, 0.f); v.y = fmaxf(v.y, 0.f);
                v.z = fmaxf(v.z, 0.f); v.w = fmaxf(v.w, 0.f);
            }
            int ch  = c4 >> 1;                               // 8-wide K chunk
            int chp = ch;                                    // natural K order
            v4bf p = { f2bf(v.x), f2bf(v.y), f2bf(v.z), f2bf(v.w) };
            *(v4bf*)(&sA[r * KB + chp * 8 + (c4 & 1) * 4]) = p;
        }
        // prefetch next A k-tile (global_prefetch_b8)
        if (k0 + KB < K) {
            int pr = rowBase + tid;
            if (pr < M) __builtin_prefetch(A + (size_t)pr * K + k0 + KB, 0, 0);
        }
        // ---- stage B tile (KB x NC), transposed to N-major bf16 ----
#pragma unroll
        for (int i = 0; i < (KB * NC / 4) / 256; ++i) {
            int idx = tid + i * 256;
            int kk  = idx / (NC / 4);
            int n4  = idx % (NC / 4);
            float4 v = *(const float4*)(B + (size_t)(k0 + kk) * NC + n4 * 4);
            sB[(n4 * 4 + 0) * KB + kk] = f2bf(v.x);
            sB[(n4 * 4 + 1) * KB + kk] = f2bf(v.y);
            sB[(n4 * 4 + 2) * KB + kk] = f2bf(v.z);
            sB[(n4 * 4 + 3) * KB + kk] = f2bf(v.w);
        }
        __syncthreads();

        // ---- A fragments: two 16x32 bf16 tiles (contiguous k = 16*khalf + s) ----
        v16bf afrag[MT];
#pragma unroll
        for (int mt = 0; mt < MT; ++mt)
            afrag[mt] = *(const v16bf*)(
                &sA[(wave * 32 + mt * 16 + m16) * KB + khalf * 16]);

        // ---- per N-tile: one B fragment feeds MT WMMAs ----
#pragma unroll
        for (int t = 0; t < NT; ++t) {
            v16bf bfrag = *(const v16bf*)(&sB[(t * 16 + m16) * KB + khalf * 16]);
#pragma unroll
            for (int mt = 0; mt < MT; ++mt)
                acc[mt][t] = __builtin_amdgcn_wmma_f32_16x16x32_bf16(
                    false, afrag[mt], false, bfrag, (short)0, acc[mt][t],
                    false, false);
        }
        __syncthreads();
    }

    // ---- epilogue: C layout = lane n, VGPR r -> row (khalf*8 + r) ----
#pragma unroll
    for (int mt = 0; mt < MT; ++mt) {
#pragma unroll
        for (int t = 0; t < NT; ++t) {
            int n    = t * 16 + m16;
            float bv = BIAS ? bias[n] : 0.0f;
#pragma unroll
            for (int r = 0; r < 8; ++r) {
                int m = rowBase + wave * 32 + mt * 16 + khalf * 8 + r;
                if (m < M) {
                    float v = acc[mt][t][r] + bv;
                    if (RELU_OUT) v = fmaxf(v, 0.f);
                    C[(size_t)m * NC + n] = v;
                }
            }
        }
    }
}

// ---------------------------------------------------------------------------
// out[i*F + f] = bias[f]  (broadcast init before scatter-accumulate)
// ---------------------------------------------------------------------------
__global__ void bias_init_kernel(float* __restrict__ out,
                                 const float* __restrict__ bias,
                                 int total4, int fmask4)
{
    int i = blockIdx.x * blockDim.x + threadIdx.x;
    if (i >= total4) return;
    ((float4*)out)[i] = ((const float4*)bias)[i & fmask4];
}

// ---------------------------------------------------------------------------
// SpMM scatter: out[row[e]] += w[e] * t[col[e]]  via hardware f32 atomics.
// One thread per (edge, float4-of-features). Gathers hit L2 (t <= 25.6MB).
// ---------------------------------------------------------------------------
template <int F>
__global__ __launch_bounds__(256) void spmm_scatter_kernel(
    const int* __restrict__ erow, const int* __restrict__ ecol,
    const float* __restrict__ ew, const float* __restrict__ t,
    float* __restrict__ out, int E)
{
    constexpr int P = F / 4;
    int gid = blockIdx.x * blockDim.x + threadIdx.x;
    int e   = gid / P;
    int f4  = gid - e * P;
    if (e >= E) return;
    int   r  = erow[e];
    int   c  = ecol[e];
    float wv = ew[e];
    float4 v = ((const float4*)(t + (size_t)c * F))[f4];
    float* o = out + (size_t)r * F + f4 * 4;
    unsafeAtomicAdd(o + 0, wv * v.x);
    unsafeAtomicAdd(o + 1, wv * v.y);
    unsafeAtomicAdd(o + 2, wv * v.z);
    unsafeAtomicAdd(o + 3, wv * v.w);
}

// ---------------------------------------------------------------------------
extern "C" void kernel_launch(void* const* d_in, const int* in_sizes, int n_in,
                              void* d_out, int out_size, void* d_ws, size_t ws_size,
                              hipStream_t stream)
{
    const float* x   = (const float*)d_in[0];
    const int*   er  = (const int*)  d_in[1];
    const int*   ec  = (const int*)  d_in[2];
    const float* ewt = (const float*)d_in[3];
    const float* W1  = (const float*)d_in[4];
    const float* b1  = (const float*)d_in[5];
    const float* W2  = (const float*)d_in[6];
    const float* b2  = (const float*)d_in[7];
    const float* W3  = (const float*)d_in[8];
    const float* b3  = (const float*)d_in[9];
    const float* P1  = (const float*)d_in[10];
    const float* pb1 = (const float*)d_in[11];
    const float* P2  = (const float*)d_in[12];
    const float* pb2 = (const float*)d_in[13];

    const int N = in_sizes[0] / 256;   // IN_F = 256
    const int E = in_sizes[1];

    float* ws0 = (float*)d_ws;                 // [N,128] GEMM output buffer
    float* ws1 = ws0 + (size_t)N * 128;        // [N,128] SpMM/activation buffer
    float* z   = (float*)d_out;                // [N,128] output 0
    float* emb = z + (size_t)N * 128;          // [N, 64] output 1

    const dim3 blk(256);
    const int  gG = (N + 255) / 256;
    auto cdiv = [](int a, int b) { return (a + b - 1) / b; };

    // ---- layer 1: t = x @ W1 ; h1 = spmm(t) + b1 (relu deferred) ----
    gemm_wmma_bf16<128, false, false, false><<<gG, blk, 0, stream>>>(x, W1, nullptr, ws0, N, 256);
    bias_init_kernel<<<cdiv(N * 128 / 4, 256), blk, 0, stream>>>(ws1, b1, N * 128 / 4, 31);
    spmm_scatter_kernel<128><<<cdiv(E * 32, 256), blk, 0, stream>>>(er, ec, ewt, ws0, ws1, E);

    // ---- layer 2: t = relu(h1) @ W2 ; h2 = spmm(t) + b2 ----
    gemm_wmma_bf16<128, true, false, false><<<gG, blk, 0, stream>>>(ws1, W2, nullptr, ws0, N, 128);
    bias_init_kernel<<<cdiv(N * 128 / 4, 256), blk, 0, stream>>>(ws1, b2, N * 128 / 4, 31);
    spmm_scatter_kernel<128><<<cdiv(E * 32, 256), blk, 0, stream>>>(er, ec, ewt, ws0, ws1, E);

    // ---- layer 3: t = relu(h2) @ W3 ; emb = spmm(t) + b3 ----
    gemm_wmma_bf16<64, true, false, false><<<gG, blk, 0, stream>>>(ws1, W3, nullptr, ws0, N, 128);
    bias_init_kernel<<<cdiv(N * 64 / 4, 256), blk, 0, stream>>>(emb, b3, N * 64 / 4, 15);
    spmm_scatter_kernel<64><<<cdiv(E * 16, 256), blk, 0, stream>>>(er, ec, ewt, ws0, emb, E);

    // ---- projection head: z = relu(emb @ P1 + pb1) @ P2 + pb2 ----
    gemm_wmma_bf16<128, false, true, true ><<<gG, blk, 0, stream>>>(emb, P1, pb1, ws1, N, 64);
    gemm_wmma_bf16<128, false, true, false><<<gG, blk, 0, stream>>>(ws1, P2, pb2, z, N, 128);
}
